// TransMILCritic_21990232555646
// MI455X (gfx1250) — matrix operand order repaired and checked by reading, not compiled, not run on hardware
//
#include <hip/hip_runtime.h>
#include <cstdint>

#define HEADS 8
#define DH    64
#define D     512
#define NTOK  16384
#define NP1   16385   // N + cls
#define PAD   255
#define NPAD  16640   // padded sequence (256*65)
#define LAND  256
#define LSEG  65
#define QKVD  1536
#define LN_EPS 1e-5f

typedef __attribute__((ext_vector_type(16))) _Float16 v16h;
typedef __attribute__((ext_vector_type(8)))  _Float16 v8h;
typedef __attribute__((ext_vector_type(8)))  float    v8f;
typedef __attribute__((ext_vector_type(2)))  float    v2f;

// ---------------------------------------------------------------------------
// Weight prep: Bh[n*K + k] = (f16) B[k*N + n]   (transpose + convert, once)
// ---------------------------------------------------------------------------
__global__ void bprep_kernel(const float* __restrict__ B, _Float16* __restrict__ Bh,
                             int K, int N) {
  int n = blockIdx.x * 256 + threadIdx.x;
  int k = blockIdx.y;
  Bh[(size_t)n * K + k] = (_Float16)B[(size_t)k * N + n];
}

// ---------------------------------------------------------------------------
// WMMA GEMM: C[M,N] = op(A[M,K](f32) @ Bh[N,K](f16,pre-transposed) + bias)
// block = 256 threads = 8 waves, each wave one 16x16 tile (2x4 tile grid)
// ---------------------------------------------------------------------------
__global__ void gemm_wmma_kernel(const float* __restrict__ A, const _Float16* __restrict__ Bh,
                                 const float* __restrict__ bias, float* __restrict__ C,
                                 int M, int N, int K, int relu, int resid) {
  int wave   = threadIdx.x >> 5;
  int lane   = threadIdx.x & 31;
  int laneHi = lane >> 4, laneLo = lane & 15;
  int row0 = (blockIdx.y * 2 + (wave >> 2)) * 16;
  int col0 = (blockIdx.x * 4 + (wave & 3)) * 16;

  v8f acc = {};
  int arow = row0 + laneLo;
  bool avalid = arow < M;
  const float* Arow = A + (size_t)(avalid ? arow : 0) * K;  // clamped: loads always safe
  int col = col0 + laneLo;
  const _Float16* Bcol = Bh + (size_t)col * K;

  if (row0 + 15 < M) {
    // ---- hot path: all 16 rows of this wave's tile valid, no masking ----
    for (int k0 = 0; k0 < K; k0 += 32) {
      const float4* ap  = (const float4*)(Arow + k0 + (laneHi << 3));
      const float4* ap2 = (const float4*)(Arow + k0 + (laneHi << 3) + 16);
      float4 ra0 = ap[0],  ra1 = ap[1];
      float4 ra2 = ap2[0], ra3 = ap2[1];
      const v8h* bp = (const v8h*)(Bcol + k0 + (laneHi << 3));
      v8h b0 = bp[0], b1 = bp[2];
      v16h a, b;
      a[0]=(_Float16)ra0.x; a[1]=(_Float16)ra0.y; a[2]=(_Float16)ra0.z; a[3]=(_Float16)ra0.w;
      a[4]=(_Float16)ra1.x; a[5]=(_Float16)ra1.y; a[6]=(_Float16)ra1.z; a[7]=(_Float16)ra1.w;
      a[8]=(_Float16)ra2.x; a[9]=(_Float16)ra2.y; a[10]=(_Float16)ra2.z; a[11]=(_Float16)ra2.w;
      a[12]=(_Float16)ra3.x; a[13]=(_Float16)ra3.y; a[14]=(_Float16)ra3.z; a[15]=(_Float16)ra3.w;
#pragma unroll
      for (int e = 0; e < 8; ++e) { b[e] = b0[e]; b[8 + e] = b1[e]; }
      acc = __builtin_amdgcn_wmma_f32_16x16x32_f16(false, a, false, b,
                                                   (short)0, acc, false, false);
    }
  } else {
    // ---- boundary path: zero-mask rows >= M ----
    float amask = avalid ? 1.f : 0.f;
    for (int k0 = 0; k0 < K; k0 += 32) {
      const float4* ap  = (const float4*)(Arow + k0 + (laneHi << 3));
      const float4* ap2 = (const float4*)(Arow + k0 + (laneHi << 3) + 16);
      float4 ra0 = ap[0],  ra1 = ap[1];
      float4 ra2 = ap2[0], ra3 = ap2[1];
      const v8h* bp = (const v8h*)(Bcol + k0 + (laneHi << 3));
      v8h b0 = bp[0], b1 = bp[2];
      v16h a, b;
      a[0]=(_Float16)(ra0.x*amask); a[1]=(_Float16)(ra0.y*amask);
      a[2]=(_Float16)(ra0.z*amask); a[3]=(_Float16)(ra0.w*amask);
      a[4]=(_Float16)(ra1.x*amask); a[5]=(_Float16)(ra1.y*amask);
      a[6]=(_Float16)(ra1.z*amask); a[7]=(_Float16)(ra1.w*amask);
      a[8]=(_Float16)(ra2.x*amask); a[9]=(_Float16)(ra2.y*amask);
      a[10]=(_Float16)(ra2.z*amask); a[11]=(_Float16)(ra2.w*amask);
      a[12]=(_Float16)(ra3.x*amask); a[13]=(_Float16)(ra3.y*amask);
      a[14]=(_Float16)(ra3.z*amask); a[15]=(_Float16)(ra3.w*amask);
#pragma unroll
      for (int e = 0; e < 8; ++e) { b[e] = b0[e]; b[8 + e] = b1[e]; }
      acc = __builtin_amdgcn_wmma_f32_16x16x32_f16(false, a, false, b,
                                                   (short)0, acc, false, false);
    }
  }

  float bv = bias ? bias[col] : 0.f;
#pragma unroll
  for (int r = 0; r < 8; ++r) {           // C: VGPR r -> M = r + 8*laneHi
    int row = row0 + r + (laneHi << 3);
    if (row < M) {
      float v = acc[r] + bv;
      if (relu) v = fmaxf(v, 0.f);
      size_t idx = (size_t)row * N + col;
      if (resid) v += C[idx];
      C[idx] = v;
    }
  }
}

// ---------------------------------------------------------------------------
// LayerNorm of (NP1 x 512) rows into padded (NPAD x 512), first PAD rows = 0
// ---------------------------------------------------------------------------
__global__ void ln_pad_kernel(const float* __restrict__ x, const float* __restrict__ g,
                              const float* __restrict__ b, float* __restrict__ out) {
  __shared__ float red[256];
  int row = blockIdx.x, t = threadIdx.x;
  float* o = out + (size_t)row * D;
  if (row < PAD) { o[t] = 0.f; o[t + 256] = 0.f; return; }
  const float* xr = x + (size_t)(row - PAD) * D;
  float a0 = xr[t], a1 = xr[t + 256];
  red[t] = a0 + a1; __syncthreads();
  for (int s = 128; s; s >>= 1) { if (t < s) red[t] += red[t + s]; __syncthreads(); }
  float mu = red[0] * (1.0f / D); __syncthreads();
  float d0 = a0 - mu, d1 = a1 - mu;
  red[t] = d0 * d0 + d1 * d1; __syncthreads();
  for (int s = 128; s; s >>= 1) { if (t < s) red[t] += red[t + s]; __syncthreads(); }
  float rinv = rsqrtf(red[0] * (1.0f / D) + LN_EPS);
  o[t]       = d0 * rinv * g[t] + b[t];
  o[t + 256] = d1 * rinv * g[t + 256] + b[t + 256];
}

// ---------------------------------------------------------------------------
// Landmarks: ql[h,m,d] = (0.125/65) * sum_j q[m*65+j, h, d]; kl similarly
// ---------------------------------------------------------------------------
__global__ void landmark_kernel(const float* __restrict__ qkv, float* __restrict__ ql,
                                float* __restrict__ kl) {
  int m = blockIdx.x, h = blockIdx.y, d = threadIdx.x;   // 64 threads
  float sq = 0.f, sk = 0.f;
  size_t base = (size_t)m * LSEG * QKVD + h * DH + d;
  for (int j = 0; j < LSEG; ++j) {
    sq += qkv[base];
    sk += qkv[base + D];
    base += QKVD;
  }
  ql[((size_t)h * LAND + m) * DH + d] = sq * (0.125f / (float)LSEG);
  kl[((size_t)h * LAND + m) * DH + d] = sk * (1.0f / (float)LSEG);
}

// ---------------------------------------------------------------------------
// a2 = softmax(ql @ kl^T), per (head, row)
// ---------------------------------------------------------------------------
__global__ void sim2_softmax_kernel(const float* __restrict__ ql, const float* __restrict__ kl,
                                    float* __restrict__ a2) {
  __shared__ float qrow[DH];
  __shared__ float red[256];
  int m = blockIdx.x, h = blockIdx.y, j = threadIdx.x;
  if (j < DH) qrow[j] = ql[((size_t)h * LAND + m) * DH + j];
  __syncthreads();
  const float* krow = kl + ((size_t)h * LAND + j) * DH;
  float s = 0.f;
#pragma unroll
  for (int d = 0; d < DH; ++d) s += qrow[d] * krow[d];
  red[j] = s; __syncthreads();
  for (int o = 128; o; o >>= 1) { if (j < o) red[j] = fmaxf(red[j], red[j + o]); __syncthreads(); }
  float mx = red[0]; __syncthreads();
  float e = __expf(s - mx);
  red[j] = e; __syncthreads();
  for (int o = 128; o; o >>= 1) { if (j < o) red[j] += red[j + o]; __syncthreads(); }
  a2[((size_t)h * LAND + m) * LAND + j] = e / red[0];
}

__global__ void zero_scal_kernel(float* scal) { if (threadIdx.x < 2) scal[threadIdx.x] = 0.f; }

// max row-sum (mode 0) and max col-sum (mode 1) of |a2| (entries positive)
__global__ void pinv_scale_kernel(const float* __restrict__ a2, float* scal) {
  __shared__ float red[256];
  int idx = blockIdx.x, h = blockIdx.y, mode = blockIdx.z, t = threadIdx.x;
  float v = (mode == 0) ? a2[((size_t)h * LAND + idx) * LAND + t]
                        : a2[((size_t)h * LAND + t) * LAND + idx];
  red[t] = v; __syncthreads();
  for (int o = 128; o; o >>= 1) { if (t < o) red[t] += red[t + o]; __syncthreads(); }
  if (t == 0) atomicMax((unsigned int*)(scal + mode), __float_as_uint(red[0]));
}

__global__ void zinit_kernel(const float* __restrict__ a2, const float* __restrict__ scal,
                             float* __restrict__ z) {
  int i = blockIdx.x, h = blockIdx.y, j = threadIdx.x;
  float s = scal[0] * scal[1];
  z[((size_t)h * LAND + j) * LAND + i] = a2[((size_t)h * LAND + i) * LAND + j] / s;
}

// ---------------------------------------------------------------------------
// pinv batched GEMM in full FP32 WMMA (16x16x4):
//   C[h] = scale * A[h] @ (ibeta*I - B[h])   (ibeta==0 -> scale * A[h] @ B[h])
// ---------------------------------------------------------------------------
__global__ void pinv_gemm_wmma_kernel(const float* __restrict__ A, const float* __restrict__ B,
                                      float* __restrict__ C, float ibeta, float scale) {
  int wave   = threadIdx.x >> 5;
  int lane   = threadIdx.x & 31;
  int laneHi = lane >> 4, laneLo = lane & 15;
  int h = blockIdx.z;
  int row0 = (blockIdx.y * 2 + (wave >> 2)) * 16;
  int col0 = (blockIdx.x * 4 + (wave & 3)) * 16;
  const float* Ah = A + (size_t)h * LAND * LAND + (size_t)(row0 + laneLo) * LAND;
  const float* Bh = B + (size_t)h * LAND * LAND + (col0 + laneLo);
  int col = col0 + laneLo;

  v8f acc = {};
  for (int k0 = 0; k0 < LAND; k0 += 4) {
    int kb = k0 + (laneHi << 1);              // element e: K = kb + e
    const float2* ap = (const float2*)(Ah + kb);
    float2 av = *ap;
    v2f a, b;
    a[0] = av.x; a[1] = av.y;
    float b0 = Bh[(size_t)kb * LAND];
    float b1 = Bh[(size_t)(kb + 1) * LAND];
    if (ibeta != 0.f) {
      b[0] = ((kb     == col) ? ibeta : 0.f) - b0;
      b[1] = ((kb + 1 == col) ? ibeta : 0.f) - b1;
    } else {
      b[0] = b0; b[1] = b1;
    }
    acc = __builtin_amdgcn_wmma_f32_16x16x4_f32(false, a, false, b,
                                                (short)0, acc, false, false);
  }
#pragma unroll
  for (int r = 0; r < 8; ++r) {
    int row = row0 + r + (laneHi << 3);
    C[((size_t)h * LAND + row) * LAND + col] = acc[r] * scale;
  }
}

// ---------------------------------------------------------------------------
// b1[h,m,:] = softmax_i(ql[h,m]·k[h,i]) @ v[h,i,:]   (streamed, wave per row)
// ---------------------------------------------------------------------------
__global__ void b1_kernel(const float* __restrict__ qkv, const float* __restrict__ ql,
                          float* __restrict__ b1) {
  __shared__ float qrow[8][DH];
  __shared__ float accS[8][DH];
  int tid = threadIdx.x, w = tid >> 5, lane = tid & 31;
  int r = blockIdx.x * 8 + w;
  int h = r >> 8, m = r & 255;
  qrow[w][lane]      = ql[((size_t)h * LAND + m) * DH + lane];
  qrow[w][lane + 32] = ql[((size_t)h * LAND + m) * DH + lane + 32];
  accS[w][lane] = 0.f; accS[w][lane + 32] = 0.f;
  __syncthreads();
  const float* kbase = qkv + D + (size_t)h * DH;
  const float* vbase = qkv + 2 * D + (size_t)h * DH;

  float lmax = -3.0e38f;
  for (int i = lane; i < NPAD; i += 32) {
    const float4* kp = (const float4*)(kbase + (size_t)i * QKVD);
    float s = 0.f;
#pragma unroll
    for (int q4 = 0; q4 < 16; ++q4) {
      float4 kv = kp[q4];
      s += qrow[w][4*q4+0]*kv.x + qrow[w][4*q4+1]*kv.y + qrow[w][4*q4+2]*kv.z + qrow[w][4*q4+3]*kv.w;
    }
    lmax = fmaxf(lmax, s);
  }
  for (int o = 16; o; o >>= 1) lmax = fmaxf(lmax, __shfl_xor(lmax, o));

  float accl[DH];
#pragma unroll
  for (int d = 0; d < DH; ++d) accl[d] = 0.f;
  float sumE = 0.f;
  for (int i = lane; i < NPAD; i += 32) {
    const float4* kp = (const float4*)(kbase + (size_t)i * QKVD);
    float s = 0.f;
#pragma unroll
    for (int q4 = 0; q4 < 16; ++q4) {
      float4 kv = kp[q4];
      s += qrow[w][4*q4+0]*kv.x + qrow[w][4*q4+1]*kv.y + qrow[w][4*q4+2]*kv.z + qrow[w][4*q4+3]*kv.w;
    }
    float e = __expf(s - lmax);
    sumE += e;
    const float4* vp = (const float4*)(vbase + (size_t)i * QKVD);
#pragma unroll
    for (int q4 = 0; q4 < 16; ++q4) {
      float4 vv = vp[q4];
      accl[4*q4+0] += e * vv.x; accl[4*q4+1] += e * vv.y;
      accl[4*q4+2] += e * vv.z; accl[4*q4+3] += e * vv.w;
    }
  }
  for (int o = 16; o; o >>= 1) sumE += __shfl_xor(sumE, o);
#pragma unroll
  for (int d = 0; d < DH; ++d) atomicAdd(&accS[w][d], accl[d]);
  __syncthreads();
  float inv = 1.0f / sumE;
  b1[((size_t)h * LAND + m) * DH + lane]      = accS[w][lane] * inv;
  b1[((size_t)h * LAND + m) * DH + lane + 32] = accS[w][lane + 32] * inv;
}

// w2 = Z @ b1
__global__ void w2_kernel(const float* __restrict__ Z, const float* __restrict__ b1,
                          float* __restrict__ w2) {
  int m = blockIdx.x, h = blockIdx.y, d = threadIdx.x;  // 64 threads
  const float* zr = Z + ((size_t)h * LAND + m) * LAND;
  const float* bh = b1 + (size_t)h * LAND * DH;
  float acc = 0.f;
  for (int k = 0; k < LAND; ++k) acc += zr[k] * bh[(size_t)k * DH + d];
  w2[((size_t)h * LAND + m) * DH + d] = acc;
}

// ---------------------------------------------------------------------------
// out[h,i,:] = softmax_m(q[i]·kl[h,m]) @ w2[h] + depthwise33x1(v)  (wave=head)
// ---------------------------------------------------------------------------
__global__ void attnout_kernel(const float* __restrict__ qkv, const float* __restrict__ kl,
                               const float* __restrict__ w2, const float* __restrict__ resw,
                               float* __restrict__ outc) {
  __shared__ float qrow[8][DH];
  __shared__ float pbuf[8][LAND];
  int i = blockIdx.x, tid = threadIdx.x;
  int h = tid >> 5, lane = tid & 31;
  qrow[h][lane]      = qkv[(size_t)i * QKVD + h * DH + lane]      * 0.125f;
  qrow[h][lane + 32] = qkv[(size_t)i * QKVD + h * DH + lane + 32] * 0.125f;
  __syncthreads();

  float s[8];
  float lmax = -3.0e38f;
#pragma unroll
  for (int t = 0; t < 8; ++t) {
    int m = lane + 32 * t;
    const float4* kp = (const float4*)(kl + ((size_t)h * LAND + m) * DH);
    float acc = 0.f;
#pragma unroll
    for (int q4 = 0; q4 < 16; ++q4) {
      float4 kv = kp[q4];
      acc += qrow[h][4*q4+0]*kv.x + qrow[h][4*q4+1]*kv.y + qrow[h][4*q4+2]*kv.z + qrow[h][4*q4+3]*kv.w;
    }
    s[t] = acc; lmax = fmaxf(lmax, acc);
  }
  for (int o = 16; o; o >>= 1) lmax = fmaxf(lmax, __shfl_xor(lmax, o));
  float lsum = 0.f;
#pragma unroll
  for (int t = 0; t < 8; ++t) { s[t] = __expf(s[t] - lmax); lsum += s[t]; pbuf[h][lane + 32 * t] = s[t]; }
  for (int o = 16; o; o >>= 1) lsum += __shfl_xor(lsum, o);
  __syncthreads();

  int d0 = lane, d1 = lane + 32;
  float o0 = 0.f, o1 = 0.f;
  const float* wh = w2 + (size_t)h * LAND * DH;
  for (int m = 0; m < LAND; ++m) {
    float p = pbuf[h][m];
    o0 += p * wh[(size_t)m * DH + d0];
    o1 += p * wh[(size_t)m * DH + d1];
  }
  float inv = 1.0f / lsum;
  o0 *= inv; o1 *= inv;

  float r0 = 0.f, r1 = 0.f;                 // depthwise conv 33x1 along tokens on v
#pragma unroll
  for (int t = 0; t < 33; ++t) {
    int ii = i + t - 16;
    if (ii >= 0 && ii < NPAD) {
      float wgt = resw[h * 33 + t];
      r0 += wgt * qkv[(size_t)ii * QKVD + 2 * D + h * DH + d0];
      r1 += wgt * qkv[(size_t)ii * QKVD + 2 * D + h * DH + d1];
    }
  }
  outc[(size_t)i * D + h * DH + d0] = o0 + r0;
  outc[(size_t)i * D + h * DH + d1] = o1 + r1;
}

// ---------------------------------------------------------------------------
// PPEG: x2 = cls | img + dw7 + dw5 + dw3 (depthwise, 128x128 image, 512 ch)
// ---------------------------------------------------------------------------
__global__ void ppeg_kernel(const float* __restrict__ x,
                            const float* __restrict__ w7, const float* __restrict__ b7,
                            const float* __restrict__ w5, const float* __restrict__ b5,
                            const float* __restrict__ w3, const float* __restrict__ b3,
                            float* __restrict__ x2) {
  int c = blockIdx.x * 256 + threadIdx.x;
  int py = blockIdx.y;
  if (py == 0) { x2[c] = x[c]; return; }
  int p = py - 1, hh = p >> 7, ww = p & 127;
  float acc = x[(size_t)py * D + c] + b7[c] + b5[c] + b3[c];
#pragma unroll
  for (int dy = -3; dy <= 3; ++dy) {
    int yy = hh + dy; if (yy < 0 || yy >= 128) continue;
#pragma unroll
    for (int dx = -3; dx <= 3; ++dx) {
      int xx = ww + dx; if (xx < 0 || xx >= 128) continue;
      float v = x[((size_t)(yy * 128 + xx) + 1) * D + c];
      acc += v * w7[(size_t)c * 49 + (dy + 3) * 7 + (dx + 3)];
      if (dy >= -2 && dy <= 2 && dx >= -2 && dx <= 2)
        acc += v * w5[(size_t)c * 25 + (dy + 2) * 5 + (dx + 2)];
      if (dy >= -1 && dy <= 1 && dx >= -1 && dx <= 1)
        acc += v * w3[(size_t)c * 9 + (dy + 1) * 3 + (dx + 1)];
    }
  }
  x2[(size_t)py * D + c] = acc;
}

// final: out = LN(x2[0]) @ fc2_w + fc2_b
__global__ void final_kernel(const float* __restrict__ x, const float* __restrict__ g,
                             const float* __restrict__ b, const float* __restrict__ w,
                             const float* __restrict__ fb, float* __restrict__ out) {
  __shared__ float red[512];
  int t = threadIdx.x;
  float v = x[t];
  red[t] = v; __syncthreads();
  for (int o = 256; o; o >>= 1) { if (t < o) red[t] += red[t + o]; __syncthreads(); }
  float mu = red[0] / (float)D; __syncthreads();
  float dv = v - mu;
  red[t] = dv * dv; __syncthreads();
  for (int o = 256; o; o >>= 1) { if (t < o) red[t] += red[t + o]; __syncthreads(); }
  float rinv = rsqrtf(red[0] / (float)D + LN_EPS); __syncthreads();
  float xn = dv * rinv * g[t] + b[t];
  red[t] = xn * w[t * 2 + 0]; __syncthreads();
  for (int o = 256; o; o >>= 1) { if (t < o) red[t] += red[t + o]; __syncthreads(); }
  float s0 = red[0]; __syncthreads();
  red[t] = xn * w[t * 2 + 1]; __syncthreads();
  for (int o = 256; o; o >>= 1) { if (t < o) red[t] += red[t + o]; __syncthreads(); }
  if (t == 0) { out[0] = s0 + fb[0]; out[1] = red[0] + fb[1]; }
}

__global__ void copy_row_kernel(const float* __restrict__ src, float* __restrict__ dst) {
  dst[threadIdx.x] = src[threadIdx.x];
}

// ---------------------------------------------------------------------------
extern "C" void kernel_launch(void* const* d_in, const int* in_sizes, int n_in,
                              void* d_out, int out_size, void* d_ws, size_t ws_size,
                              hipStream_t stream) {
  (void)in_sizes; (void)n_in; (void)out_size;
  const float* h      = (const float*)d_in[0];
  const float* fc1_w  = (const float*)d_in[1];
  const float* fc1_b  = (const float*)d_in[2];
  const float* cls    = (const float*)d_in[3];
  const float* l1_g   = (const float*)d_in[4];
  const float* l1_b   = (const float*)d_in[5];
  const float* l1_qkv = (const float*)d_in[6];
  const float* l1_ow  = (const float*)d_in[7];
  const float* l1_ob  = (const float*)d_in[8];
  const float* l1_rw  = (const float*)d_in[9];
  const float* w7     = (const float*)d_in[10];
  const float* b7     = (const float*)d_in[11];
  const float* w5     = (const float*)d_in[12];
  const float* b5     = (const float*)d_in[13];
  const float* w3     = (const float*)d_in[14];
  const float* b3     = (const float*)d_in[15];
  const float* l2_g   = (const float*)d_in[16];
  const float* l2_b   = (const float*)d_in[17];
  const float* l2_qkv = (const float*)d_in[18];
  const float* l2_ow  = (const float*)d_in[19];
  const float* l2_ob  = (const float*)d_in[20];
  const float* l2_rw  = (const float*)d_in[21];
  const float* ng     = (const float*)d_in[22];
  const float* nb     = (const float*)d_in[23];
  const float* fc2w   = (const float*)d_in[24];
  const float* fc2b   = (const float*)d_in[25];
  float* out = (float*)d_out;

  float* ws = (float*)d_ws;
  size_t off = 0;
  auto alloc = [&](size_t n) { float* p = ws + off; off += n; return p; };
  float* xbuf  = alloc((size_t)NP1 * D);
  float* x2buf = alloc((size_t)NP1 * D);
  float* lnbuf = alloc((size_t)NPAD * D);        // doubles as attn-concat buffer
  float* qkv   = alloc((size_t)NPAD * QKVD);
  float* ql    = alloc((size_t)HEADS * LAND * DH);
  float* kl    = alloc((size_t)HEADS * LAND * DH);
  float* a2    = alloc((size_t)HEADS * LAND * LAND);
  float* z0    = alloc((size_t)HEADS * LAND * LAND);
  float* z1    = alloc((size_t)HEADS * LAND * LAND);
  float* xz    = alloc((size_t)HEADS * LAND * LAND);
  float* tb    = alloc((size_t)HEADS * LAND * LAND);
  float* ub    = alloc((size_t)HEADS * LAND * LAND);
  float* b1    = alloc((size_t)HEADS * LAND * DH);
  float* w2b   = alloc((size_t)HEADS * LAND * DH);
  float* scal  = alloc(16);
  _Float16* bh = (_Float16*)alloc((size_t)2048 * 512 / 2);  // f16 weight scratch (max K*N)
  if (off * sizeof(float) > ws_size) return;     // insufficient scratch

  auto pinv_gemm = [&](const float* Aa, const float* Bb, float* Cc, float ib, float sc) {
    pinv_gemm_wmma_kernel<<<dim3(LAND / 64, LAND / 32, HEADS), 256, 0, stream>>>(Aa, Bb, Cc, ib, sc);
  };

  auto attention = [&](float* x, const float* lng, const float* lnb2,
                       const float* qkvw, const float* ow, const float* ob,
                       const float* rw) {
    ln_pad_kernel<<<NPAD, 256, 0, stream>>>(x, lng, lnb2, lnbuf);
    bprep_kernel<<<dim3(QKVD / 256, D), 256, 0, stream>>>(qkvw, bh, D, QKVD);
    gemm_wmma_kernel<<<dim3(QKVD / 64, NPAD / 32), 256, 0, stream>>>(
        lnbuf, bh, nullptr, qkv, NPAD, QKVD, D, 0, 0);
    landmark_kernel<<<dim3(LAND, HEADS), 64, 0, stream>>>(qkv, ql, kl);
    sim2_softmax_kernel<<<dim3(LAND, HEADS), 256, 0, stream>>>(ql, kl, a2);
    zero_scal_kernel<<<1, 32, 0, stream>>>(scal);
    pinv_scale_kernel<<<dim3(LAND, HEADS, 2), 256, 0, stream>>>(a2, scal);
    zinit_kernel<<<dim3(LAND, HEADS), 256, 0, stream>>>(a2, scal, z0);
    float* Z = z0; float* Zn = z1;
    for (int it = 0; it < 6; ++it) {
      pinv_gemm(a2, Z, xz, 0.f, 1.f);
      pinv_gemm(xz, xz, tb, 7.f, 1.f);
      pinv_gemm(xz, tb, ub, 15.f, 1.f);
      pinv_gemm(Z, ub, Zn, 13.f, 0.25f);
      float* tmp = Z; Z = Zn; Zn = tmp;
    }
    b1_kernel<<<LAND, 256, 0, stream>>>(qkv, ql, b1);
    w2_kernel<<<dim3(LAND, HEADS), 64, 0, stream>>>(Z, b1, w2b);
    attnout_kernel<<<NPAD, 256, 0, stream>>>(qkv, kl, w2b, rw, lnbuf);
    bprep_kernel<<<dim3(D / 256, D), 256, 0, stream>>>(ow, bh, D, D);
    gemm_wmma_kernel<<<dim3(D / 64, (NP1 + 31) / 32), 256, 0, stream>>>(
        lnbuf + (size_t)PAD * D, bh, ob, x, NP1, D, D, 0, 1);
  };

  copy_row_kernel<<<1, D, 0, stream>>>(cls, xbuf);
  bprep_kernel<<<dim3(D / 256, 2048), 256, 0, stream>>>(fc1_w, bh, 2048, D);
  gemm_wmma_kernel<<<dim3(D / 64, NTOK / 32), 256, 0, stream>>>(
      h, bh, fc1_b, xbuf + D, NTOK, D, 2048, 1, 0);
  attention(xbuf, l1_g, l1_b, l1_qkv, l1_ow, l1_ob, l1_rw);
  ppeg_kernel<<<dim3(2, NP1), 256, 0, stream>>>(xbuf, w7, b7, w5, b5, w3, b3, x2buf);
  attention(x2buf, l2_g, l2_b, l2_qkv, l2_ow, l2_ob, l2_rw);
  final_kernel<<<1, D, 0, stream>>>(x2buf, ng, nb, fc2w, fc2b, out);
}
